// BalancedEILayer_89189290869302
// MI455X (gfx1250) — matrix operand order repaired and checked by reading, not compiled, not run on hardware
//
#include <hip/hip_runtime.h>

// ---------------------------------------------------------------------------
// Balanced E/I recurrent layer, CDNA5 (gfx1250) WMMA bf16 implementation.
// Concatenated formulation:
//   s = [e | i]  (B x 2560), r = relu(s)
//   G[k][n]  = Dale-law weight block matrix (2560 x 2560), so rec = r @ G
//   GX[k][n] = input weights (1024 x 2560), drive = bf16(x) @ GX
// GEMM staging uses GLOBAL_LOAD_ASYNC_TO_LDS_B128 (ASYNCcnt), double-buffered
// LDS tiles, branch-free software pipeline (2x unrolled, peeled tail).
// Requires K % 64 == 0 (holds: K = 1024 and 2560).
// ---------------------------------------------------------------------------

typedef __attribute__((ext_vector_type(16))) __bf16       v16bf;
typedef __attribute__((ext_vector_type(8)))  float        v8f;
typedef __attribute__((ext_vector_type(4)))  unsigned int v4u;

#define NE_    2048
#define NI_    512
#define NC_    2560      // NE + NI
#define BATCH_ 4096
#define DIM_   1024
#define TSTEPS 5
#define STEP_NB 5120     // (BATCH*NC) / (256*8)

__device__ __forceinline__ unsigned short f2bf(float f) {
  unsigned int u = __float_as_uint(f);
  u += 0x7fffu + ((u >> 16) & 1u);          // round-to-nearest-even
  return (unsigned short)(u >> 16);
}
__device__ __forceinline__ float softplus_f(float x) {
  return (x > 20.f) ? x : log1pf(expf(x));
}

// One wave-wide async copy: each lane moves 16B global -> LDS (ASYNCcnt).
#define ASYNC_LD128(ldsoff, gaddr)                                       \
  asm volatile("global_load_async_to_lds_b128 %0, %1, off"               \
               :: "v"(ldsoff), "v"(gaddr) : "memory")
#define WAIT_ASYNC_LE4() asm volatile("s_wait_asynccnt 0x4" ::: "memory")
#define WAIT_ASYNC_0()   asm volatile("s_wait_asynccnt 0x0" ::: "memory")

// ---------------- prep kernels ----------------------------------------------

__global__ void prep_G(const float* __restrict__ tEE, const float* __restrict__ tEI,
                       const float* __restrict__ tIE, const float* __restrict__ tII,
                       unsigned short* __restrict__ G) {
  int idx = blockIdx.x * blockDim.x + threadIdx.x;
  if (idx >= NC_ * NC_) return;
  int k = idx / NC_, n = idx % NC_;
  float v;
  if (n < NE_) {
    if (k < NE_) v =  softplus_f(tEE[n * NE_ + k]) * (1.0f / NE_);
    else         v = -softplus_f(tEI[n * NI_ + (k - NE_)]) * (1.0f / NI_);
  } else {
    int nn = n - NE_;
    if (k < NE_) v =  softplus_f(tIE[nn * NE_ + k]) * (1.0f / NE_);
    else         v = -softplus_f(tII[nn * NI_ + (k - NE_)]) * (1.0f / NI_);
  }
  G[idx] = f2bf(v);
}

__global__ void prep_GX(const float* __restrict__ WXEw, const float* __restrict__ WXIw,
                        unsigned short* __restrict__ GX) {
  int idx = blockIdx.x * blockDim.x + threadIdx.x;
  if (idx >= DIM_ * NC_) return;
  int k = idx / NC_, n = idx % NC_;
  float v = (n < NE_) ? WXEw[n * DIM_ + k] : WXIw[(n - NE_) * DIM_ + k];
  GX[idx] = f2bf(v);
}

__global__ void prep_X(const float* __restrict__ x, unsigned short* __restrict__ xb) {
  int idx = blockIdx.x * blockDim.x + threadIdx.x;
  if (idx >= BATCH_ * DIM_) return;
  xb[idx] = f2bf(x[idx]);
}

__global__ void prep_misc(const float* __restrict__ WXEb, const float* __restrict__ WXIb,
                          const float* __restrict__ be,  const float* __restrict__ bi,
                          float* __restrict__ S, float* __restrict__ bias) {
  int idx = blockIdx.x * blockDim.x + threadIdx.x;
  if (idx < BATCH_ * NC_) S[idx] = 0.f;
  if (idx < NC_)
    bias[idx] = (idx < NE_) ? (WXEb[idx] + be[idx]) : (WXIb[idx - NE_] + bi[idx - NE_]);
}

// ---------------- bf16 WMMA GEMM:  C[M][N] = A[M][K] * B[K][N] --------------
// Block = 256 threads (8 wave32), 128x128 tile, K-step 32, double-buffered
// LDS fed by global_load_async_to_lds_b128; branch-free pipeline.
// Wave (wr,wc) owns a 64x32 sub-tile = 4x2 WMMA 16x16 tiles.

#define LDA_P 40
#define LDB_P 136

__global__ __launch_bounds__(256) void gemm_bf16(const unsigned short* __restrict__ A,
                                                 const unsigned short* __restrict__ Bm,
                                                 float* __restrict__ C,
                                                 int M, int N, int K) {
  __shared__ __align__(16) unsigned short As[2][128 * LDA_P];  // [row][k], padded
  __shared__ __align__(16) unsigned short Bs[2][32 * LDB_P];   // [k][n],  padded

  const int tid  = threadIdx.x;
  const int lane = tid & 31, wave = tid >> 5;
  const int wr = wave >> 2, wc = wave & 3;       // wr: 0..1, wc: 0..3
  const int half = lane >> 4, l16 = lane & 15;
  const int n0 = blockIdx.x * 128;
  const int m0 = blockIdx.y * 128;

  // Per-thread staging assignment: 2 A chunks + 2 B chunks of 16B each.
  const int ia0r = tid >> 2,         ia0c = (tid & 3) << 3;
  const int ia1r = (tid + 256) >> 2, ia1c = ia0c;
  const int ib0r = tid >> 4,         ib0c = (tid & 15) << 3;
  const int ib1r = (tid + 256) >> 4, ib1c = ib0c;

  const unsigned short* gA0 = A  + (size_t)(m0 + ia0r) * K + ia0c;
  const unsigned short* gA1 = A  + (size_t)(m0 + ia1r) * K + ia1c;
  const unsigned short* gB0 = Bm + (size_t)ib0r * N + n0 + ib0c;
  const unsigned short* gB1 = Bm + (size_t)ib1r * N + n0 + ib1c;
  const size_t bStep = (size_t)32 * N;   // elements per K-step for B rows

  unsigned lA0[2], lA1[2], lB0[2], lB1[2];
  #pragma unroll
  for (int b = 0; b < 2; b++) {
    lA0[b] = (unsigned)(size_t)(const void*)&As[b][ia0r * LDA_P + ia0c];
    lA1[b] = (unsigned)(size_t)(const void*)&As[b][ia1r * LDA_P + ia1c];
    lB0[b] = (unsigned)(size_t)(const void*)&Bs[b][ib0r * LDB_P + ib0c];
    lB1[b] = (unsigned)(size_t)(const void*)&Bs[b][ib1r * LDB_P + ib1c];
  }

  v8f acc[4][2] = {};

  // Issue one K-step's 4 async copies into buffer b and advance pointers.
  auto stageBuf = [&](int b) {
    ASYNC_LD128(lA0[b], (unsigned long long)(size_t)gA0);
    ASYNC_LD128(lA1[b], (unsigned long long)(size_t)gA1);
    ASYNC_LD128(lB0[b], (unsigned long long)(size_t)gB0);
    ASYNC_LD128(lB1[b], (unsigned long long)(size_t)gB1);
    gA0 += 32; gA1 += 32; gB0 += bStep; gB1 += bStep;
  };

  // 8 WMMAs on buffer b (fragment layouts per ISA 7.12.2).
  auto computeBuf = [&](int b) {
    union Frag { v16bf f; v4u u[2]; } af[4], bfr[2];
    #pragma unroll
    for (int nt = 0; nt < 2; nt++) {           // B: lane%16 = N col; K halves (table order)
      const unsigned short* p = &Bs[b][(half * 8) * LDB_P + wc * 32 + nt * 16 + l16];
      #pragma unroll
      for (int s = 0; s < 8; s++) {
        bfr[nt].f[s]     = __builtin_bit_cast(__bf16, p[s * LDB_P]);
        bfr[nt].f[s + 8] = __builtin_bit_cast(__bf16, p[(16 + s) * LDB_P]);
      }
    }
    #pragma unroll
    for (int mt = 0; mt < 4; mt++) {           // A: lane = M row; K halves
      const unsigned short* p = &As[b][(wr * 64 + mt * 16 + l16) * LDA_P + half * 8];
      af[mt].u[0] = *(const v4u*)(p);
      af[mt].u[1] = *(const v4u*)(p + 16);
    }
    #pragma unroll
    for (int mt = 0; mt < 4; mt++)
      #pragma unroll
      for (int nt = 0; nt < 2; nt++)
        acc[mt][nt] = __builtin_amdgcn_wmma_f32_16x16x32_bf16(
            false, af[mt].f, false, bfr[nt].f, (short)0, acc[mt][nt], false, false);
  };

  const int nk = K >> 5;                // even (K % 64 == 0)

  // Software pipeline: prologue, branch-free steady state (2x unrolled), tail.
  stageBuf(0);
  for (int it = 0; it < nk - 2; it += 2) {
    stageBuf(1);
    WAIT_ASYNC_LE4();
    __syncthreads();
    computeBuf(0);
    __syncthreads();
    stageBuf(0);
    WAIT_ASYNC_LE4();
    __syncthreads();
    computeBuf(1);
    __syncthreads();
  }
  stageBuf(1);
  WAIT_ASYNC_LE4();
  __syncthreads();
  computeBuf(0);
  __syncthreads();
  WAIT_ASYNC_0();
  __syncthreads();
  computeBuf(1);

  // C/D layout: VGPR r -> row M = half*8 + r, lane -> col N = l16
  #pragma unroll
  for (int mt = 0; mt < 4; mt++)
    #pragma unroll
    for (int nt = 0; nt < 2; nt++)
      #pragma unroll
      for (int r = 0; r < 8; r++) {
        int row = m0 + wr * 64 + mt * 16 + half * 8 + r;
        int col = n0 + wc * 32 + nt * 16 + l16;
        C[(size_t)row * N + col] = acc[mt][nt][r];
      }
}

// ---------------- fused state update + partial reductions -------------------

__global__ __launch_bounds__(256) void step_k(const float* __restrict__ rec,
                                              const float* __restrict__ drive,
                                              const float* __restrict__ bias,
                                              float* __restrict__ s,
                                              unsigned short* __restrict__ rb,
                                              float* __restrict__ part,
                                              int zero_rec) {
  __shared__ float redA[256], redB[256];
  const int tid = threadIdx.x;
  const size_t base = (size_t)blockIdx.x * (256 * 8) + tid;
  const float wE = 1.0f / ((float)BATCH_ * (float)NE_);
  const float wI = 1.0f / ((float)BATCH_ * (float)NI_);
  float sa = 0.f, sb = 0.f;
  #pragma unroll
  for (int i = 0; i < 8; i++) {
    size_t idx = base + (size_t)i * 256;
    int n = (int)(idx % NC_);
    float w  = (n < NE_) ? wE : wI;
    float sv = s[idx];
    float re = fmaxf(sv, 0.f);
    sa += re * re * w;                                     // ac term (f32 relu)
    float bal = drive[idx] + (zero_rec ? 0.f : rec[idx]);  // rec + drive
    sb += bal * bal * w;                                   // bc term
    float snew = 0.8f * sv + 0.2f * (bal + bias[n]);
    s[idx] = snew;
    rb[idx] = f2bf(fmaxf(snew, 0.f));                      // r_{t+1} for GEMM
  }
  redA[tid] = sa; redB[tid] = sb;
  __syncthreads();
  for (int off = 128; off > 0; off >>= 1) {
    if (tid < off) { redA[tid] += redA[tid + off]; redB[tid] += redB[tid + off]; }
    __syncthreads();
  }
  if (tid == 0) {
    part[blockIdx.x] = redA[0];
    part[STEP_NB + blockIdx.x] = redB[0];
  }
}

// ---------------- epilogue --------------------------------------------------

__global__ void out_e(const float* __restrict__ s, float* __restrict__ out) {
  int idx = blockIdx.x * blockDim.x + threadIdx.x;
  if (idx >= BATCH_ * NE_) return;
  int b = idx / NE_, n = idx % NE_;
  out[idx] = fmaxf(s[(size_t)b * NC_ + n], 0.f);
}

__global__ __launch_bounds__(256) void fin_k(const float* __restrict__ part,
                                             float* __restrict__ out) {
  __shared__ float redA[256], redB[256];
  const int tid = threadIdx.x;
  float sa = 0.f, sb = 0.f;
  for (int t = 0; t < TSTEPS; t++) {
    const float* p = part + (size_t)t * 2 * STEP_NB;
    for (int i = tid; i < STEP_NB; i += 256) { sa += p[i]; sb += p[STEP_NB + i]; }
  }
  redA[tid] = sa; redB[tid] = sb;
  __syncthreads();
  for (int off = 128; off > 0; off >>= 1) {
    if (tid < off) { redA[tid] += redA[tid + off]; redB[tid] += redB[tid + off]; }
    __syncthreads();
  }
  if (tid == 0) {
    out[(size_t)BATCH_ * NE_]     = redA[0] / (float)TSTEPS;
    out[(size_t)BATCH_ * NE_ + 1] = redB[0] / (float)TSTEPS;
  }
}

// ---------------- host ------------------------------------------------------

extern "C" void kernel_launch(void* const* d_in, const int* in_sizes, int n_in,
                              void* d_out, int out_size, void* d_ws, size_t ws_size,
                              hipStream_t stream) {
  const float* x    = (const float*)d_in[0];
  const float* tEE  = (const float*)d_in[1];
  const float* tEI  = (const float*)d_in[2];
  const float* tIE  = (const float*)d_in[3];
  const float* tII  = (const float*)d_in[4];
  const float* WXEw = (const float*)d_in[5];
  const float* WXEb = (const float*)d_in[6];
  const float* WXIw = (const float*)d_in[7];
  const float* WXIb = (const float*)d_in[8];
  const float* be   = (const float*)d_in[9];
  const float* bi   = (const float*)d_in[10];
  float* out = (float*)d_out;

  char* ws = (char*)d_ws;
  size_t off = 0;
  auto alloc = [&](size_t bytes) -> void* {
    void* p = ws + off;
    off = (off + bytes + 255) & ~(size_t)255;
    return p;
  };
  unsigned short* G    = (unsigned short*)alloc((size_t)NC_ * NC_ * 2);     // 13.1 MB
  unsigned short* GX   = (unsigned short*)alloc((size_t)DIM_ * NC_ * 2);    //  5.2 MB
  unsigned short* XB   = (unsigned short*)alloc((size_t)BATCH_ * DIM_ * 2); //  8.4 MB
  unsigned short* RB   = (unsigned short*)alloc((size_t)BATCH_ * NC_ * 2);  // 21.0 MB
  float*          S    = (float*)alloc((size_t)BATCH_ * NC_ * 4);           // 41.9 MB
  float*          DR   = (float*)alloc((size_t)BATCH_ * NC_ * 4);           // 41.9 MB
  float*          RC   = (float*)alloc((size_t)BATCH_ * NC_ * 4);           // 41.9 MB
  float*          BIAS = (float*)alloc((size_t)NC_ * 4);
  float*          PART = (float*)alloc((size_t)TSTEPS * 2 * STEP_NB * 4);

  // Prep: weights (softplus + Dale + scale) -> bf16, x -> bf16, zero state.
  prep_G   <<<(NC_ * NC_ + 255) / 256, 256, 0, stream>>>(tEE, tEI, tIE, tII, G);
  prep_GX  <<<(DIM_ * NC_ + 255) / 256, 256, 0, stream>>>(WXEw, WXIw, GX);
  prep_X   <<<(BATCH_ * DIM_ + 255) / 256, 256, 0, stream>>>(x, XB);
  prep_misc<<<(BATCH_ * NC_ + 255) / 256, 256, 0, stream>>>(WXEb, WXIb, be, bi, S, BIAS);

  // drive = bf16(x) @ GX   (M=4096, N=2560, K=1024)
  dim3 gg(NC_ / 128, BATCH_ / 128);
  gemm_bf16<<<gg, 256, 0, stream>>>(XB, GX, DR, BATCH_, NC_, DIM_);

  // T recurrence steps: rec = r @ G (M=4096, N=2560, K=2560), then fused update
  for (int t = 0; t < TSTEPS; t++) {
    if (t > 0)
      gemm_bf16<<<gg, 256, 0, stream>>>(RB, G, RC, BATCH_, NC_, NC_);
    step_k<<<STEP_NB, 256, 0, stream>>>(RC, DR, BIAS, S, RB,
                                        PART + (size_t)t * 2 * STEP_NB,
                                        (t == 0) ? 1 : 0);
  }

  // Outputs: relu(e), ac/T, bc/T
  out_e<<<(BATCH_ * NE_ + 255) / 256, 256, 0, stream>>>(S, out);
  fin_k<<<1, 256, 0, stream>>>(PART, out);
}
